// Attention_68341519614426
// MI455X (gfx1250) — matrix operand order repaired and checked by reading, not compiled, not run on hardware
//
#include <hip/hip_runtime.h>
#include <hip/hip_bf16.h>

// ---------------------------------------------------------------------------
// Fused QKV projection + multi-head attention for MI455X (gfx1250, wave32).
// B=4, N=2048, D=1024, H=16, pd=64.  All matrix math on v_wmma_f32_16x16x32_bf16.
// Pipeline: (1) convert x,W to bf16   (2) WMMA GEMM qkv = x@W^T + b, scatter
// q/k/vT to workspace   (3) WMMA flash attention with double-buffered
// global_load_async_to_lds staging of K/V tiles.
// ---------------------------------------------------------------------------

typedef __attribute__((ext_vector_type(16))) __bf16 v16bf;
typedef __attribute__((ext_vector_type(8)))  float  v8f;

#define WMMA_BF16(a, b, c)                                                     \
  __builtin_amdgcn_wmma_f32_16x16x32_bf16(false, (a), false, (b), (short)0,    \
                                          (c), false, false)

union FragBF {
  v16bf v;
  unsigned int u[8];
  int4 q[2];
};

// RNE f32->bf16 pair pack (used only outside hot loops).
static __device__ __forceinline__ unsigned int pk_bf16_rne(float a, float b) {
  unsigned int ua = __float_as_uint(a);
  unsigned int ub = __float_as_uint(b);
  ua = (ua + 0x7fffu + ((ua >> 16) & 1u)) >> 16;
  ub = (ub + 0x7fffu + ((ub >> 16) & 1u)) >> 16;
  return ua | (ub << 16);
}

static __device__ __forceinline__ unsigned short cvt_bf16(float a) {
  unsigned int ua = __float_as_uint(a);
  return (unsigned short)((ua + 0x7fffu + ((ua >> 16) & 1u)) >> 16);
}

// Truncating f32->bf16 pair pack: single v_perm_b32.
static __device__ __forceinline__ unsigned int pk_bf16_trunc(float a, float b) {
  return __builtin_amdgcn_perm(__float_as_uint(b), __float_as_uint(a),
                               0x07060302u);
}

// Async copy of 16 bytes global -> LDS (per lane), tracked by ASYNCcnt.
static __device__ __forceinline__ void async_b128_to_lds(
    unsigned lds_byte_off, const void* gptr) {
  asm volatile("global_load_async_to_lds_b128 %0, %1, off"
               :
               : "v"(lds_byte_off), "v"((unsigned long long)(size_t)gptr)
               : "memory");
}

static __device__ __forceinline__ void wait_asynccnt0() {
#if __has_builtin(__builtin_amdgcn_s_wait_asynccnt)
  __builtin_amdgcn_s_wait_asynccnt(0);
#else
  asm volatile("s_wait_asynccnt 0x0" ::: "memory");
#endif
}

// Generic shared-memory pointer -> LDS byte offset (low 32 bits of the
// shared-aperture address are the LDS byte address, per ISA flat addressing).
static __device__ __forceinline__ unsigned lds_off(const void* p) {
  return (unsigned)(size_t)p;
}

// A-matrix fragment (16x32, bf16): lane m (0-15) = row m, K elems k0+0..15 contiguous;
// lanes 16-31 = same rows, K elems k0+16..31.
static __device__ __forceinline__ v16bf load_a_bf16(const unsigned short* base,
                                                    int ld, int row0, int k0,
                                                    int lane) {
  int r = row0 + (lane & 15);
  const unsigned short* p = base + (size_t)r * ld + k0 + ((lane >> 4) & 1) * 16;
  FragBF f;
  f.q[0] = *(const int4*)(p);
  f.q[1] = *(const int4*)(p + 8);
  return f.v;
}

// B-matrix fragment (32x16, bf16) loaded from B^T stored row-major (rows = N cols of B):
// lane n (0-15) = column n, K elems k0+0..15 contiguous; lanes 16-31: K elems k0+16..31.
static __device__ __forceinline__ v16bf load_b_bf16(const unsigned short* base,
                                                    int ld, int col0, int k0,
                                                    int lane) {
  int r = col0 + (lane & 15);
  const unsigned short* p = base + (size_t)r * ld + k0 + ((lane >> 4) & 1) * 16;
  FragBF f;
  f.q[0] = *(const int4*)(p);
  f.q[1] = *(const int4*)(p + 8);
  return f.v;
}

// ---------------------------------------------------------------------------
// Kernel 0: f32 -> bf16 bulk convert (RNE), 4 elements per thread.
// ---------------------------------------------------------------------------
__global__ __launch_bounds__(256) void cvt_kernel(const float* __restrict__ src,
                                                  unsigned int* __restrict__ dst,
                                                  int n4) {
  int i = blockIdx.x * 256 + threadIdx.x;
  if (i >= n4) return;
  float4 v = ((const float4*)src)[i];
  uint2 o;
  o.x = pk_bf16_rne(v.x, v.y);
  o.y = pk_bf16_rne(v.z, v.w);
  ((uint2*)dst)[i] = o;
}

// ---------------------------------------------------------------------------
// Kernel 1: qkv = xb @ wb^T + b, scattered to q(b,h,n,p), k(b,h,n,p), vT(b,h,p,n)
// M = B*N = 8192, Nout = 3072, K = 1024.  Each wave: one 32x32 tile.
// ---------------------------------------------------------------------------
static __device__ __forceinline__ void store_tile(v8f c, int tm16, int tn16,
                                                  int lane, const float* bias,
                                                  unsigned short* qb,
                                                  unsigned short* kb,
                                                  unsigned short* vt) {
  int col = tn16 + (lane & 15);
  float bv = bias[col];
  int cid = col >> 10;        // 0=q, 1=k, 2=v
  int rem = col & 1023;
  int h = rem >> 6;
  int pp = rem & 63;
  int rowbase = tm16 + ((lane >> 4) & 1) * 8;
#pragma unroll
  for (int j = 0; j < 8; ++j) {
    int row = rowbase + j;                 // 0..8191
    int b = row >> 11;
    int n = row & 2047;
    unsigned short val = cvt_bf16(c[j] + bv);
    size_t bh = (size_t)(b * 16 + h);
    if (cid == 0)
      qb[(bh * 2048 + n) * 64 + pp] = val;
    else if (cid == 1)
      kb[(bh * 2048 + n) * 64 + pp] = val;
    else
      vt[(bh * 64 + pp) * 2048 + n] = val;
  }
}

__global__ __launch_bounds__(256) void qkv_proj_kernel(
    const unsigned short* __restrict__ xb, const unsigned short* __restrict__ wb,
    const float* __restrict__ bias, unsigned short* __restrict__ qb,
    unsigned short* __restrict__ kb, unsigned short* __restrict__ vt) {
  const int lane = threadIdx.x & 31;
  const int wave = threadIdx.x >> 5;
  int job = blockIdx.x * 8 + wave;   // 24576 wave jobs
  int tn = (job % 96) * 32;          // output column base (0..3071)
  int tm = (job / 96) * 32;          // output row base    (0..8191)

  v8f c00 = {0.f, 0.f, 0.f, 0.f, 0.f, 0.f, 0.f, 0.f};
  v8f c01 = c00, c10 = c00, c11 = c00;

  for (int k0 = 0; k0 < 1024; k0 += 32) {
    v16bf a0 = load_a_bf16(xb, 1024, tm, k0, lane);
    v16bf a1 = load_a_bf16(xb, 1024, tm + 16, k0, lane);
    v16bf b0 = load_b_bf16(wb, 1024, tn, k0, lane);
    v16bf b1 = load_b_bf16(wb, 1024, tn + 16, k0, lane);
    c00 = WMMA_BF16(a0, b0, c00);
    c01 = WMMA_BF16(a0, b1, c01);
    c10 = WMMA_BF16(a1, b0, c10);
    c11 = WMMA_BF16(a1, b1, c11);
  }
  store_tile(c00, tm, tn, lane, bias, qb, kb, vt);
  store_tile(c01, tm, tn + 16, lane, bias, qb, kb, vt);
  store_tile(c10, tm + 16, tn, lane, bias, qb, kb, vt);
  store_tile(c11, tm + 16, tn + 16, lane, bias, qb, kb, vt);
}

// ---------------------------------------------------------------------------
// Kernel 2: flash attention.  Wave = 16 q rows; workgroup = 128 q rows of one
// (b,h).  S^T = K*Q^T so q lives in the lane index (softmax needs only xor-16
// combines); out^T accumulated as V^T * P^T so alpha rescale is lane-uniform.
// K/V tiles double-buffered in LDS via global_load_async_to_lds_b128.
// ---------------------------------------------------------------------------
#define K_LDS_LD 72   // 64 + 8 ushort pad (144B row stride, 16B aligned)
#define V_LDS_LD 40   // 32 + 8 ushort pad (80B row stride, 16B aligned)

__global__ __launch_bounds__(256) void attn_kernel(
    const unsigned short* __restrict__ qb, const unsigned short* __restrict__ kb,
    const unsigned short* __restrict__ vt, float* __restrict__ out) {
  __shared__ unsigned short smemK[2][32 * K_LDS_LD];  // 32 kv rows x 64 pd
  __shared__ unsigned short smemV[2][64 * V_LDS_LD];  // 64 pd rows x 32 kv

  const int tid = threadIdx.x;
  const int lane = tid & 31;
  const int wave = tid >> 5;
  const int bh = blockIdx.x >> 4;      // 0..63  (b*16+h)
  const int qt = blockIdx.x & 15;
  const int q0 = qt * 128 + wave * 16;
  const int hi = (lane >> 4) & 1;

  const unsigned short* Q = qb + (size_t)bh * 2048 * 64;
  const unsigned short* K = kb + (size_t)bh * 2048 * 64;
  const unsigned short* V = vt + (size_t)bh * 64 * 2048;

  // Per-thread staging slots (one 16B async op each for K and V tiles).
  const int krow = tid >> 3, kchk = tid & 7;   // K: 32 rows x 8 chunks
  const int vrow = tid >> 2, vchk = tid & 3;   // V: 64 rows x 4 chunks
  const unsigned kdst0 = lds_off(&smemK[0][krow * K_LDS_LD + kchk * 8]);
  const unsigned kdst1 = lds_off(&smemK[1][krow * K_LDS_LD + kchk * 8]);
  const unsigned vdst0 = lds_off(&smemV[0][vrow * V_LDS_LD + vchk * 8]);
  const unsigned vdst1 = lds_off(&smemV[1][vrow * V_LDS_LD + vchk * 8]);

  // Q as B-operand (persistent), K dim = pd (64) in two 32-chunks.
  v16bf bq0 = load_b_bf16(Q, 64, q0, 0, lane);
  v16bf bq1 = load_b_bf16(Q, 64, q0, 32, lane);

  // Prologue: stage kv tile 0 into buffer 0.
  async_b128_to_lds(kdst0, K + (size_t)krow * 64 + kchk * 8);
  async_b128_to_lds(vdst0, V + (size_t)vrow * 2048 + vchk * 8);
  wait_asynccnt0();
  __syncthreads();

  v8f z = {0.f, 0.f, 0.f, 0.f, 0.f, 0.f, 0.f, 0.f};
  v8f acc0 = z, acc1 = z, acc2 = z, acc3 = z;   // out^T, pd blocks 0..3
  float m = -3.0e38f, l = 0.f;

  for (int kv = 0; kv < 2048; kv += 32) {
    const int buf = (kv >> 5) & 1;
    // Stage next tile into the other buffer (overlaps with compute below).
    if (kv + 32 < 2048) {
      async_b128_to_lds(buf ? kdst0 : kdst1,
                        K + (size_t)(kv + 32 + krow) * 64 + kchk * 8);
      async_b128_to_lds(buf ? vdst0 : vdst1,
                        V + (size_t)vrow * 2048 + (kv + 32) + vchk * 8);
    }
    const unsigned short* sK = smemK[buf];
    const unsigned short* sV = smemV[buf];

    // S^T tiles: rows kv..kv+15 (s0) and kv+16..kv+31 (s1), cols = 16 q's.
    v8f s0 = z, s1 = z;
    s0 = WMMA_BF16(load_a_bf16(sK, K_LDS_LD, 0, 0, lane), bq0, s0);
    s0 = WMMA_BF16(load_a_bf16(sK, K_LDS_LD, 0, 32, lane), bq1, s0);
    s1 = WMMA_BF16(load_a_bf16(sK, K_LDS_LD, 16, 0, lane), bq0, s1);
    s1 = WMMA_BF16(load_a_bf16(sK, K_LDS_LD, 16, 32, lane), bq1, s1);

    // Online softmax over the 32 kv scores of q = lane&15 (split lane/lane^16).
    float tmax = s0[0];
#pragma unroll
    for (int j = 1; j < 8; ++j) tmax = fmaxf(tmax, s0[j]);
#pragma unroll
    for (int j = 0; j < 8; ++j) tmax = fmaxf(tmax, s1[j]);
    tmax = fmaxf(tmax, __shfl_xor(tmax, 16, 32));
    float mn = fmaxf(m, tmax);
    float alpha = __expf(m - mn);
    m = mn;

    float p0[8], p1[8], ts = 0.f;
#pragma unroll
    for (int j = 0; j < 8; ++j) {
      p0[j] = __expf(s0[j] - mn);
      p1[j] = __expf(s1[j] - mn);
      ts += p0[j] + p1[j];
    }
    ts += __shfl_xor(ts, 16, 32);
    l = l * alpha + ts;
#pragma unroll
    for (int j = 0; j < 8; ++j) {
      acc0[j] *= alpha; acc1[j] *= alpha; acc2[j] *= alpha; acc3[j] *= alpha;
    }

    // Repack P into B-operand layout (K=kv, N=q).  Low lanes own kv 0..7 and
    // need partner's kv 8..15; high lanes own kv 24..31 and need partner's 16..23.
    float ex[8];
#pragma unroll
    for (int j = 0; j < 8; ++j) ex[j] = __shfl_xor(hi ? p0[j] : p1[j], 16, 32);
    float lo[8], hh[8];
#pragma unroll
    for (int j = 0; j < 8; ++j) {
      lo[j] = hi ? ex[j] : p0[j];
      hh[j] = hi ? p1[j] : ex[j];
    }
    FragBF fp;
    fp.u[0] = pk_bf16_trunc(lo[0], lo[1]); fp.u[1] = pk_bf16_trunc(lo[2], lo[3]);
    fp.u[2] = pk_bf16_trunc(lo[4], lo[5]); fp.u[3] = pk_bf16_trunc(lo[6], lo[7]);
    fp.u[4] = pk_bf16_trunc(hh[0], hh[1]); fp.u[5] = pk_bf16_trunc(hh[2], hh[3]);
    fp.u[6] = pk_bf16_trunc(hh[4], hh[5]); fp.u[7] = pk_bf16_trunc(hh[6], hh[7]);
    v16bf bp = fp.v;

    // out^T += V^T(pd-block x 32kv) * P^T
    acc0 = WMMA_BF16(load_a_bf16(sV, V_LDS_LD, 0, 0, lane), bp, acc0);
    acc1 = WMMA_BF16(load_a_bf16(sV, V_LDS_LD, 16, 0, lane), bp, acc1);
    acc2 = WMMA_BF16(load_a_bf16(sV, V_LDS_LD, 32, 0, lane), bp, acc2);
    acc3 = WMMA_BF16(load_a_bf16(sV, V_LDS_LD, 48, 0, lane), bp, acc3);

    // Next-tile staging complete + all waves done reading this buffer.
    wait_asynccnt0();
    __syncthreads();
  }

  float inv = 1.0f / l;
  int b = bh >> 4, h = bh & 15;
  int n = q0 + (lane & 15);
  int prow = hi * 8;
  // Faithful buggy reshape: out[b][h*N*pd + n*pd + p]
  float* ob = out + (size_t)b * 2048 * 1024 + (size_t)h * 2048 * 64 +
              (size_t)n * 64;
  v8f accs[4] = {acc0, acc1, acc2, acc3};
#pragma unroll
  for (int blk = 0; blk < 4; ++blk) {
    float* p = ob + blk * 16 + prow;
    v8f a = accs[blk];
    *(float4*)(p) = make_float4(a[0] * inv, a[1] * inv, a[2] * inv, a[3] * inv);
    *(float4*)(p + 4) =
        make_float4(a[4] * inv, a[5] * inv, a[6] * inv, a[7] * inv);
  }
}

// ---------------------------------------------------------------------------
extern "C" void kernel_launch(void* const* d_in, const int* in_sizes, int n_in,
                              void* d_out, int out_size, void* d_ws,
                              size_t ws_size, hipStream_t stream) {
  const float* x = (const float*)d_in[0];     // (4,2048,1024)
  const float* W = (const float*)d_in[1];     // (3072,1024)
  const float* bias = (const float*)d_in[2];  // (3072,)
  float* out = (float*)d_out;                 // (4,2048,1024)

  const size_t QKV_ELEMS = (size_t)4 * 16 * 2048 * 64;  // 8,388,608 per tensor
  const size_t X_ELEMS = (size_t)4 * 2048 * 1024;       // 8,388,608
  const size_t W_ELEMS = (size_t)3072 * 1024;           // 3,145,728

  unsigned short* qb = (unsigned short*)d_ws;
  unsigned short* kb = qb + QKV_ELEMS;
  unsigned short* vt = kb + QKV_ELEMS;
  unsigned short* xb = vt + QKV_ELEMS;
  unsigned short* wb = xb + X_ELEMS;

  // (0) f32 -> bf16 pre-convert (bandwidth-trivial: ~44 MB @ 23.3 TB/s)
  cvt_kernel<<<dim3((unsigned)(X_ELEMS / 4 / 256)), dim3(256), 0, stream>>>(
      x, (unsigned int*)xb, (int)(X_ELEMS / 4));
  cvt_kernel<<<dim3((unsigned)(W_ELEMS / 4 / 256)), dim3(256), 0, stream>>>(
      W, (unsigned int*)wb, (int)(W_ELEMS / 4));

  // (1) 24576 wave-tiles (32x32) / 8 waves per block = 3072 blocks
  qkv_proj_kernel<<<dim3(3072), dim3(256), 0, stream>>>(xb, wb, bias, qb, kb,
                                                        vt);
  // (2) 64 (b,h) * 16 q-tiles of 128 rows = 1024 blocks
  attn_kernel<<<dim3(1024), dim3(256), 0, stream>>>(qb, kb, vt, out);
}
